// BeamSearch_29420525977761
// MI455X (gfx1250) — matrix operand order repaired and checked by reading, not compiled, not run on hardware
//
#include <hip/hip_runtime.h>
#include <math.h>

// ---------------- problem constants ----------------
#define BB   32
#define KK   5
#define SS   64
#define HH   768
#define VV   30522
#define TT   40
#define SEPT 102
#define NN   (BB * KK)      // 160 decode rows
#define H2   (2 * HH)       // 1536
#define H3   (3 * HH)       // 2304
#define VP   30528          // V padded to multiple of 64

// ---------------- CDNA5 WMMA types ----------------
typedef __attribute__((ext_vector_type(16))) __bf16 v16bf;
typedef __attribute__((ext_vector_type(8)))  float  v8f;
typedef __attribute__((ext_vector_type(4)))  unsigned int u32x4;

union AFrag { v16bf v; unsigned int d[8]; };
union BFrag { v16bf v; u32x4 q[2]; };

__device__ __forceinline__ unsigned short f2bf(float x) {
    unsigned u = __float_as_uint(x);
    unsigned r = u + 0x7FFFu + ((u >> 16) & 1u);   // round-to-nearest-even
    return (unsigned short)(r >> 16);
}

// ---------------- weight conversion kernels ----------------
__global__ void to_bf16_kernel(const float* __restrict__ in,
                               unsigned short* __restrict__ out, int n) {
    int i = blockIdx.x * 256 + threadIdx.x;
    if (i < n) out[i] = f2bf(in[i]);
}

// rows padded with zeros up to rowsPad (for Wo)
__global__ void to_bf16_padrows_kernel(const float* __restrict__ in,
                                       unsigned short* __restrict__ out,
                                       int rows, int cols, int rowsPad) {
    int i = blockIdx.x * 256 + threadIdx.x;
    int total = rowsPad * cols;
    if (i >= total) return;
    int r = i / cols, c = i % cols;
    out[i] = (r < rows) ? f2bf(in[(size_t)r * cols + c]) : (unsigned short)0;
}

// in [rows,cols] f32 -> out [cols,rows] bf16  (for Wa, Ua)
__global__ void transpose_bf16_kernel(const float* __restrict__ in,
                                      unsigned short* __restrict__ out,
                                      int rows, int cols) {
    int i = blockIdx.x * 256 + threadIdx.x;
    if (i >= rows * cols) return;
    int r = i / cols, c = i % cols;
    out[(size_t)c * rows + r] = f2bf(in[i]);
}

// ---------------- generic bf16 WMMA GEMM ----------------
// C[M,N] = A[M,K] * Bw[N,K]^T (+ bias[N])
// A row-major bf16 (lda); Bw row-major bf16 [N rows, K cols] (ldb), buffer
// padded to Npad = align16(N) rows, and Npad MUST be a multiple of 64.
// M multiple of 16, K multiple of 32. Wave tile: 16 rows x 64 cols
// (4 accumulators); block = 4 waves = 256 cols. The wave column base is
// CLAMPED (branch-free) instead of guarded, so the K-loop is straight-line
// and EXEC is all-ones at every WMMA (ISA requirement).
__global__ __launch_bounds__(128)
void wmma_gemm_bf16(const unsigned short* __restrict__ A, int lda,
                    const unsigned short* __restrict__ Bw, int ldb,
                    float* __restrict__ C, int ldc,
                    const float* __restrict__ bias,
                    int M, int Ncols, int Kdim) {
    const int lane = threadIdx.x & 31;
    const int wave = threadIdx.x >> 5;
    const int mbase = blockIdx.y * 16;
    const int Npad = (Ncols + 15) & ~15;          // callers guarantee %64 == 0
    int colblock = blockIdx.x * 256 + wave * 64;
    if (colblock + 64 > Npad) colblock = Npad - 64;   // clamp, no divergence later

    const int arow = mbase + (lane & 15);
    const int aoff = (lane >> 4) * 8;          // K offset for this half-wave (A)
    const int bn   = lane & 15;                // output column within subtile (B)
    const int bk   = (lane >> 4) * 8;          // K offset for this half-wave (B)

    const unsigned short* arowptr = A + (size_t)arow * lda + aoff;
    const unsigned short* bp0 = Bw + (size_t)(colblock +  0 + bn) * ldb + bk;
    const unsigned short* bp1 = Bw + (size_t)(colblock + 16 + bn) * ldb + bk;
    const unsigned short* bp2 = Bw + (size_t)(colblock + 32 + bn) * ldb + bk;
    const unsigned short* bp3 = Bw + (size_t)(colblock + 48 + bn) * ldb + bk;

    v8f acc0 = {}, acc1 = {}, acc2 = {}, acc3 = {};

    for (int k0 = 0; k0 < Kdim; k0 += 32) {
        AFrag a;
        const unsigned short* ap = arowptr + k0;
#pragma unroll
        for (int d = 0; d < 8; ++d) {
            int kk = 2 * d + ((d >= 4) ? 8 : 0);       // K pairs 0,2,4,6,16,18,20,22
            a.d[d] = *(const unsigned int*)(ap + kk);
        }
        BFrag b0, b1, b2, b3;
        b0.q[0] = *(const u32x4*)(bp0 + k0);
        b0.q[1] = *(const u32x4*)(bp0 + k0 + 16);
        b1.q[0] = *(const u32x4*)(bp1 + k0);
        b1.q[1] = *(const u32x4*)(bp1 + k0 + 16);
        b2.q[0] = *(const u32x4*)(bp2 + k0);
        b2.q[1] = *(const u32x4*)(bp2 + k0 + 16);
        b3.q[0] = *(const u32x4*)(bp3 + k0);
        b3.q[1] = *(const u32x4*)(bp3 + k0 + 16);
        acc0 = __builtin_amdgcn_wmma_f32_16x16x32_bf16(
            false, a.v, false, b0.v, (short)0, acc0, false, false);
        acc1 = __builtin_amdgcn_wmma_f32_16x16x32_bf16(
            false, a.v, false, b1.v, (short)0, acc1, false, false);
        acc2 = __builtin_amdgcn_wmma_f32_16x16x32_bf16(
            false, a.v, false, b2.v, (short)0, acc2, false, false);
        acc3 = __builtin_amdgcn_wmma_f32_16x16x32_bf16(
            false, a.v, false, b3.v, (short)0, acc3, false, false);
    }

    const int rbase = mbase + ((lane >> 4) * 8);
#pragma unroll
    for (int s = 0; s < 4; ++s) {
        int col = colblock + s * 16 + (lane & 15);
        v8f acc = (s == 0) ? acc0 : (s == 1) ? acc1 : (s == 2) ? acc2 : acc3;
        if (col < Ncols) {
            float bv = bias ? bias[col] : 0.f;
#pragma unroll
            for (int r = 0; r < 8; ++r) {
                int row = rbase + r;
                if (row < M) C[(size_t)row * ldc + col] = acc[r] + bv;
            }
        }
    }
}

// ---------------- per-step kernels ----------------
// gather hidden by backpointer; step 0: zero hidden, SEP token
__global__ __launch_bounds__(256)
void prepare_kernel(int first, float* __restrict__ h_f,
                    unsigned short* __restrict__ h_bf,
                    const float* __restrict__ h_post,
                    const int* __restrict__ maps, int* __restrict__ toks) {
    int n = blockIdx.x;
    int b = n / KK;
    if (first) {
        if (threadIdx.x == 0) toks[n] = SEPT;
        for (int j = threadIdx.x; j < HH; j += 256) {
            h_f[(size_t)n * HH + j] = 0.f;
            h_bf[(size_t)n * HH + j] = 0;
        }
    } else {
        int p = maps[n];
        const float* srcp = h_post + (size_t)(b * KK + p) * HH;
        for (int j = threadIdx.x; j < HH; j += 256) {
            float x = srcp[j];
            h_f[(size_t)n * HH + j] = x;
            h_bf[(size_t)n * HH + j] = f2bf(x);
        }
    }
}

// attention scores + softmax + context + embedding gather -> x_bf [N, 2H]
__global__ __launch_bounds__(256)
void attention_kernel(const float* __restrict__ srcWa,
                      const float* __restrict__ src,
                      const float* __restrict__ hUa,
                      const float* __restrict__ va,
                      const float* __restrict__ E,
                      const int* __restrict__ toks,
                      unsigned short* __restrict__ x_bf) {
    int n = blockIdx.x;
    int b = n / KK;
    int tid = threadIdx.x, lane = tid & 31, wave = tid >> 5;
    __shared__ float sh_h[HH];
    __shared__ float sh_a[SS];

    for (int j = tid; j < HH; j += 256) sh_h[j] = hUa[(size_t)n * HH + j];
    __syncthreads();

    for (int s = wave; s < SS; s += 8) {
        const float* sw = srcWa + (size_t)(b * SS + s) * HH;
        float p = 0.f;
        for (int j = lane; j < HH; j += 32)
            p += tanhf(sw[j] + sh_h[j]) * va[j];
        for (int off = 16; off > 0; off >>= 1) p += __shfl_xor(p, off, 32);
        if (lane == 0) sh_a[s] = p;
    }
    __syncthreads();

    if (wave == 0) {   // softmax over S=64 by one wave32
        float e0 = sh_a[lane], e1 = sh_a[lane + 32];
        float m = fmaxf(e0, e1);
        for (int off = 16; off > 0; off >>= 1) m = fmaxf(m, __shfl_xor(m, off, 32));
        float x0 = expf(e0 - m), x1 = expf(e1 - m);
        float sum = x0 + x1;
        for (int off = 16; off > 0; off >>= 1) sum += __shfl_xor(sum, off, 32);
        float inv = 1.f / sum;
        sh_a[lane] = x0 * inv;
        sh_a[lane + 32] = x1 * inv;
    }
    __syncthreads();

    for (int j = tid; j < HH; j += 256) {     // ctx
        float acc = 0.f;
        const float* sb = src + (size_t)(b * SS) * HH + j;
#pragma unroll 4
        for (int s = 0; s < SS; ++s) acc += sh_a[s] * sb[(size_t)s * HH];
        x_bf[(size_t)n * H2 + HH + j] = f2bf(acc);
    }
    int tk = toks[n];                          // embedding
    for (int j = tid; j < HH; j += 256)
        x_bf[(size_t)n * H2 + j] = f2bf(E[(size_t)tk * HH + j]);
}

// GRU gates + new hidden
__global__ __launch_bounds__(256)
void gru_kernel(const float* __restrict__ gx, const float* __restrict__ gh,
                const float* __restrict__ b_ih, const float* __restrict__ b_hh,
                const float* __restrict__ h_f, float* __restrict__ h_post,
                unsigned short* __restrict__ h2_bf) {
    int n = blockIdx.x;
    for (int j = threadIdx.x; j < HH; j += 256) {
        float xr = gx[(size_t)n * H3 + j] + b_ih[j];
        float hr = gh[(size_t)n * H3 + j] + b_hh[j];
        float xz = gx[(size_t)n * H3 + HH + j] + b_ih[HH + j];
        float hz = gh[(size_t)n * H3 + HH + j] + b_hh[HH + j];
        float xn = gx[(size_t)n * H3 + 2 * HH + j] + b_ih[2 * HH + j];
        float hn = gh[(size_t)n * H3 + 2 * HH + j] + b_hh[2 * HH + j];
        float r = 1.f / (1.f + expf(-(xr + hr)));
        float z = 1.f / (1.f + expf(-(xz + hz)));
        float nn = tanhf(xn + r * hn);
        float h2 = (1.f - z) * nn + z * h_f[(size_t)n * HH + j];
        h_post[(size_t)n * HH + j] = h2;
        h2_bf[(size_t)n * HH + j] = f2bf(h2);
    }
}

// per-row max and log-sum-exp over vocab
__global__ __launch_bounds__(256)
void row_lse_kernel(const float* __restrict__ logits, int ldl,
                    float* __restrict__ rmax, float* __restrict__ rlse) {
    int n = blockIdx.x, tid = threadIdx.x;
    __shared__ float sh[256];
    const float* row = logits + (size_t)n * ldl;
    float m = -3.402823e38f;
    for (int v = tid; v < VV; v += 256) m = fmaxf(m, row[v]);
    sh[tid] = m; __syncthreads();
    for (int off = 128; off > 0; off >>= 1) {
        if (tid < off) sh[tid] = fmaxf(sh[tid], sh[tid + off]);
        __syncthreads();
    }
    m = sh[0]; __syncthreads();
    float s = 0.f;
    for (int v = tid; v < VV; v += 256) s += expf(row[v] - m);
    sh[tid] = s; __syncthreads();
    for (int off = 128; off > 0; off >>= 1) {
        if (tid < off) sh[tid] += sh[tid + off];
        __syncthreads();
    }
    if (tid == 0) { rmax[n] = m; rlse[n] = logf(sh[0]); }
}

// beam selection: top-K over K*V candidates per batch row (step 0: over V, beam 0)
__global__ __launch_bounds__(512)
void select_kernel(int step,
                   const float* __restrict__ logits, int ldl,
                   const float* __restrict__ rmax, const float* __restrict__ rlse,
                   float* __restrict__ scores, float* __restrict__ ended,
                   float* __restrict__ lengths, int* __restrict__ toks,
                   int* __restrict__ maps, int* __restrict__ toks_hist,
                   int* __restrict__ maps_hist) {
    int b = blockIdx.x, tid = threadIdx.x;
    __shared__ float bv[512];
    __shared__ int   bt[512];
    __shared__ float o_cand[KK];
    __shared__ int   o_idx[KK];

    float den[KK], sc[KK], en[KK], lenNew[KK];
#pragma unroll
    for (int k = 0; k < KK; ++k) {
        float e = (step == 0) ? 0.f : ended[b * KK + k];
        float l = (step == 0) ? 0.f : lengths[b * KK + k];
        float lN = l + ((e == 0.f) ? 1.f : 0.f);
        lenNew[k] = lN;
        en[k] = e;
        sc[k] = (step == 0) ? 0.f : scores[b * KK + k];
        den[k] = (step == 0) ? 1.f : ((k < KK - 1) ? powf(lN, 0.7f) : 1.f);
    }
    int kmax = (step == 0) ? 1 : KK;

    float tn[KK], tc[KK];
    int   ti[KK];
#pragma unroll
    for (int i = 0; i < KK; ++i) { tn[i] = -3.0e38f; tc[i] = -3.0e38f; ti[i] = 0; }

    for (int k = 0; k < kmax; ++k) {
        int rown = b * KK + k;
        float rm = rmax[rown], rl = rlse[rown];
        const float* row = logits + (size_t)rown * ldl;
        float ek = en[k], sk = sc[k], dk = den[k];
        for (int v = tid; v < VV; v += 512) {
            float lp = (ek != 0.f) ? ((v == SEPT) ? 0.f : -1e9f)
                                   : (row[v] - rm - rl);
            float cand = lp + sk;
            float nrm = cand / dk;
            if (nrm > tn[KK - 1]) {
                int pos = KK - 1;
                while (pos > 0 && nrm > tn[pos - 1]) {
                    tn[pos] = tn[pos - 1]; tc[pos] = tc[pos - 1]; ti[pos] = ti[pos - 1];
                    --pos;
                }
                tn[pos] = nrm; tc[pos] = cand; ti[pos] = k * VV + v;
            }
        }
    }

    int head = 0;
    for (int sel = 0; sel < KK; ++sel) {       // pop best across 512 local lists
        bv[tid] = (head < KK) ? tn[head] : -3.0e38f;
        bt[tid] = tid;
        __syncthreads();
        for (int off = 256; off > 0; off >>= 1) {
            if (tid < off && bv[tid + off] > bv[tid]) {
                bv[tid] = bv[tid + off]; bt[tid] = bt[tid + off];
            }
            __syncthreads();
        }
        int w = bt[0];
        if (tid == w) { o_cand[sel] = tc[head]; o_idx[sel] = ti[head]; ++head; }
        __syncthreads();
    }

    if (tid < KK) {
        int k = tid;
        int idx = o_idx[k];
        float cand = o_cand[k];
        int ntok = idx % VV;
        int nmap = (step == 0) ? 0 : (idx / VV);
        float nend = (step == 0) ? 0.f : ((ntok == SEPT) ? 1.f : en[k]);
        float nlen = (step == 0) ? 0.f : lenNew[k];
        scores[b * KK + k] = cand;
        toks[b * KK + k] = ntok;
        maps[b * KK + k] = nmap;
        ended[b * KK + k] = nend;
        lengths[b * KK + k] = nlen;
        toks_hist[step * NN + b * KK + k] = ntok;
        maps_hist[step * NN + b * KK + k] = nmap;
    }
}

// trace back best beam; tokens written as float (output dtype)
__global__ void backtrace_kernel(const float* __restrict__ scores,
                                 const int* __restrict__ toks_hist,
                                 const int* __restrict__ maps_hist,
                                 float* __restrict__ out) {
    int b = threadIdx.x;
    if (b >= BB) return;
    float bestv = -3.0e38f; int best = 0;
    for (int k = 0; k < KK; ++k) {
        float s = scores[b * KK + k];
        if (s > bestv) { bestv = s; best = k; }
    }
    int idx = best;
    for (int t = TT - 1; t >= 0; --t) {
        out[b * TT + t] = (float)toks_hist[t * NN + b * KK + idx];
        idx = maps_hist[t * NN + b * KK + idx];
    }
}

// ---------------- host orchestration ----------------
extern "C" void kernel_launch(void* const* d_in, const int* in_sizes, int n_in,
                              void* d_out, int out_size, void* d_ws, size_t ws_size,
                              hipStream_t stream) {
    (void)in_sizes; (void)n_in; (void)out_size; (void)ws_size;
    const float* src  = (const float*)d_in[0];
    const float* E    = (const float*)d_in[1];
    const float* Wa   = (const float*)d_in[2];
    const float* Ua   = (const float*)d_in[3];
    const float* va   = (const float*)d_in[4];
    const float* Wih  = (const float*)d_in[5];
    const float* Whh  = (const float*)d_in[6];
    const float* b_ih = (const float*)d_in[7];
    const float* b_hh = (const float*)d_in[8];
    const float* Wo   = (const float*)d_in[9];
    const float* bo   = (const float*)d_in[10];

    size_t off = 0;
    char* base = (char*)d_ws;
    auto carve = [&](size_t bytes) -> char* {
        char* r = base + off;
        off += (bytes + 255) & ~(size_t)255;
        return r;
    };
    unsigned short* wo_bf  = (unsigned short*)carve((size_t)VP * HH * 2);
    unsigned short* wih_bf = (unsigned short*)carve((size_t)H3 * H2 * 2);
    unsigned short* whh_bf = (unsigned short*)carve((size_t)H3 * HH * 2);
    unsigned short* uaT_bf = (unsigned short*)carve((size_t)HH * HH * 2);
    unsigned short* waT_bf = (unsigned short*)carve((size_t)HH * HH * 2);
    unsigned short* src_bf = (unsigned short*)carve((size_t)BB * SS * HH * 2);
    float* srcWa  = (float*)carve((size_t)BB * SS * HH * 4);
    float* h_f    = (float*)carve((size_t)NN * HH * 4);
    unsigned short* h_bf  = (unsigned short*)carve((size_t)NN * HH * 2);
    float* h_post = (float*)carve((size_t)NN * HH * 4);
    unsigned short* h2_bf = (unsigned short*)carve((size_t)NN * HH * 2);
    unsigned short* x_bf  = (unsigned short*)carve((size_t)NN * H2 * 2);
    float* ghatt  = (float*)carve((size_t)NN * HH * 4);
    float* gh_f   = (float*)carve((size_t)NN * H3 * 4);
    float* gx_f   = (float*)carve((size_t)NN * H3 * 4);
    float* logits = (float*)carve((size_t)NN * VP * 4);
    float* rmax   = (float*)carve(NN * 4);
    float* rlse   = (float*)carve(NN * 4);
    float* scores = (float*)carve(NN * 4);
    float* ended  = (float*)carve(NN * 4);
    float* lengths= (float*)carve(NN * 4);
    int* toks      = (int*)carve(NN * 4);
    int* maps      = (int*)carve(NN * 4);
    int* toks_hist = (int*)carve((size_t)TT * NN * 4);
    int* maps_hist = (int*)carve((size_t)TT * NN * 4);

    // ---- one-time weight prep (bf16) ----
    {
        int n = VP * HH;
        to_bf16_padrows_kernel<<<(n + 255) / 256, 256, 0, stream>>>(Wo, wo_bf, VV, HH, VP);
        n = H3 * H2;
        to_bf16_kernel<<<(n + 255) / 256, 256, 0, stream>>>(Wih, wih_bf, n);
        n = H3 * HH;
        to_bf16_kernel<<<(n + 255) / 256, 256, 0, stream>>>(Whh, whh_bf, n);
        n = BB * SS * HH;
        to_bf16_kernel<<<(n + 255) / 256, 256, 0, stream>>>(src, src_bf, n);
        n = HH * HH;
        transpose_bf16_kernel<<<(n + 255) / 256, 256, 0, stream>>>(Wa, waT_bf, HH, HH);
        transpose_bf16_kernel<<<(n + 255) / 256, 256, 0, stream>>>(Ua, uaT_bf, HH, HH);
        // srcWa = src @ Wa  (loop-invariant)
        dim3 g((HH + 255) / 256, (BB * SS) / 16);
        wmma_gemm_bf16<<<g, 128, 0, stream>>>(src_bf, HH, waT_bf, HH,
                                              srcWa, HH, nullptr,
                                              BB * SS, HH, HH);
    }

    // ---- T sequential decode steps ----
    for (int t = 0; t < TT; ++t) {
        prepare_kernel<<<NN, 256, 0, stream>>>(t == 0 ? 1 : 0, h_f, h_bf,
                                               h_post, maps, toks);
        {   // hUa = h @ Ua
            dim3 g((HH + 255) / 256, NN / 16);
            wmma_gemm_bf16<<<g, 128, 0, stream>>>(h_bf, HH, uaT_bf, HH,
                                                  ghatt, HH, nullptr, NN, HH, HH);
        }
        {   // gh = h @ W_hh.T
            dim3 g((H3 + 255) / 256, NN / 16);
            wmma_gemm_bf16<<<g, 128, 0, stream>>>(h_bf, HH, whh_bf, HH,
                                                  gh_f, H3, nullptr, NN, H3, HH);
        }
        attention_kernel<<<NN, 256, 0, stream>>>(srcWa, src, ghatt, va, E, toks, x_bf);
        {   // gx = x @ W_ih.T
            dim3 g((H3 + 255) / 256, NN / 16);
            wmma_gemm_bf16<<<g, 128, 0, stream>>>(x_bf, H2, wih_bf, H2,
                                                  gx_f, H3, nullptr, NN, H3, H2);
        }
        gru_kernel<<<NN, 256, 0, stream>>>(gx_f, gh_f, b_ih, b_hh, h_f, h_post, h2_bf);
        {   // logits = h2 @ Wo.T + bo   (47 MB bf16 weight stream; resident in L2)
            dim3 g((VP + 255) / 256, NN / 16);
            wmma_gemm_bf16<<<g, 128, 0, stream>>>(h2_bf, HH, wo_bf, HH,
                                                  logits, VP, bo, NN, VV, HH);
        }
        row_lse_kernel<<<NN, 256, 0, stream>>>(logits, VP, rmax, rlse);
        select_kernel<<<BB, 512, 0, stream>>>(t, logits, VP, rmax, rlse,
                                              scores, ended, lengths, toks, maps,
                                              toks_hist, maps_hist);
    }

    backtrace_kernel<<<1, 32, 0, stream>>>(scores, toks_hist, maps_hist, (float*)d_out);
}
